// SSRA_7103875907592
// MI455X (gfx1250) — matrix operand order repaired and checked by reading, not compiled, not run on hardware
//
#include <hip/hip_runtime.h>
#include <hip/hip_bf16.h>

typedef __attribute__((ext_vector_type(16))) __bf16 v16bf;
typedef __attribute__((ext_vector_type(8)))  __bf16 v8bf;
typedef __attribute__((ext_vector_type(8)))  float  v8f;
typedef __attribute__((ext_vector_type(4)))  unsigned int u32x4;
typedef __attribute__((ext_vector_type(8)))  int    i32x8;
typedef __attribute__((ext_vector_type(4)))  int    i32x4;

#define Bn   16
#define Cn   32
#define Hn   128
#define Wn   128
#define Rn   8
#define OUTn 32
#define HWn  (Hn * Wn)

#if __has_builtin(__builtin_amdgcn_tensor_load_to_lds) && \
    __has_builtin(__builtin_amdgcn_s_wait_tensorcnt)
#define USE_TDM 1
#else
#define USE_TDM 0
#endif

// ---- fragment packing helpers (16-bit A-matrix 16x32, contiguous k = 16*(lane/16) + slot) ----
__device__ __forceinline__ int frag_lane(int row, int kk) { return row + 16 * (kk >> 4); }
__device__ __forceinline__ int frag_slot(int kk)          { return kk & 15; }

#if USE_TDM
// Issue one 2D TDM load: 3 rows of tile_w bf16 elements, global row stride
// row_stride elements, into contiguous LDS at lds_addr. (D# per ISA ch.8)
// Toolchain uses the 6-arg builtin: (g0 x4u, g1 x8i, g2 x4i, g3 x4i, x8i, cpol)
__device__ __forceinline__ void tdm_load_2d(unsigned lds_addr, const void* gaddr,
                                            unsigned tile_w, unsigned row_stride) {
    unsigned long long ga = (unsigned long long)gaddr;
    u32x4 g0;
    g0[0] = 1u;                                              // count=1, user mode
    g0[1] = lds_addr;                                        // LDS byte address
    g0[2] = (unsigned)ga;                                    // global_addr[31:0]
    g0[3] = (unsigned)((ga >> 32) & 0x1FFFFFFu) | (2u << 30);// addr[56:32] | type=2
    i32x8 g1;
    g1[0] = (1 << 16);                    // data_size = 1 (2 bytes)
    g1[1] = (int)(0xFFFFu << 16);         // tensor_dim0 = 0x7FFFFFFF (no clip) lo16
    g1[2] = (int)(0x7FFFu | (0xFFFFu << 16)); // dim0 hi16 | tensor_dim1 lo16
    g1[3] = (int)(0x7FFFu | (tile_w << 16));  // dim1 hi16 | tile_dim0
    g1[4] = 3;                            // tile_dim1 = 3 rows, tile_dim2 = 0
    g1[5] = (int)row_stride;              // tensor_dim0_stride[31:0]
    g1[6] = 0;                            // stride0 hi | stride1 lo
    g1[7] = 0;
    i32x4 g2 = {0, 0, 0, 0}, g3 = {0, 0, 0, 0};
    i32x8 g4 = {0, 0, 0, 0, 0, 0, 0, 0};
    __builtin_amdgcn_tensor_load_to_lds(g0, g1, g2, g3, g4, 0);
}
#endif

// =====================================================================
// k0: pack fusion-conv weights w_f[32][64][3][3] -> bf16 A-fragments.
//     K permutation: kt = kidx*2 + (c2>>5), kk = c2&31
// =====================================================================
__global__ void k0_pack_wf(const float* __restrict__ w_f, __bf16* __restrict__ wf_frag) {
    int t = threadIdx.x;
    int m = t & 31, part = t >> 5;
    for (int kq = part * 72; kq < part * 72 + 72; ++kq) {
        int c2 = kq / 9, kidx = kq % 9;
        float v = w_f[(m * 64 + c2) * 9 + kidx];
        int kt = kidx * 2 + (c2 >> 5), kk = c2 & 31;
        int idx = (((m >> 4) * 18 + kt) * 32 + frag_lane(m & 15, kk)) * 16 + frag_slot(kk);
        wf_frag[idx] = (__bf16)v;
    }
}

// =====================================================================
// k1: adaptive pool -> sigmoid FC -> grouped 1x1 -> dynamic kernels.
//     K permutation: kt = kidx, kk = cc.  grid=B, block=288
// =====================================================================
__global__ void k1_gen(const float* __restrict__ input,
                       const float* __restrict__ w1, const float* __restrict__ b1,
                       const float* __restrict__ w2, const float* __restrict__ b2,
                       __bf16* __restrict__ kern_frag) {
    __shared__ float pooled[Cn][9];
    __shared__ float gbuf[64][9];
    int b = blockIdx.x, t = threadIdx.x;

    if (t < Cn * 9) {
        int c = t / 9, bin = t % 9, by = bin / 3, bx = bin % 3;
        int hs = (by * Hn) / 3, he = ((by + 1) * Hn + 2) / 3;
        int ws = (bx * Wn) / 3, we = ((bx + 1) * Wn + 2) / 3;
        const float* p = input + ((size_t)b * Cn + c) * HWn;
        float s = 0.f;
        for (int y = hs; y < he; ++y)
            for (int x = ws; x < we; ++x) s += p[y * Wn + x];
        pooled[c][bin] = s / (float)((he - hs) * (we - ws));
    }
    __syncthreads();

    for (int idx = t; idx < 64 * 9; idx += blockDim.x) {
        int o = idx / 9, bin = idx % 9;
        float a = b1[o];
        for (int c = 0; c < Cn; ++c) a += pooled[c][bin] * w1[o * Cn + c];
        gbuf[o][bin] = 1.f / (1.f + __expf(-a));
    }
    __syncthreads();

    if (t < 256) {
        int m = t, r = m >> 5, j = m & 31;
        for (int cc = 0; cc < Cn; ++cc) {
            int oP = j * 32 + cc;
            float bias = b2[r * 1024 + oP];
            int lane = frag_lane(m & 15, cc), slot = frag_slot(cc);
            for (int kidx = 0; kidx < 9; ++kidx) {
                float acc = bias;
                for (int i = 0; i < Rn; ++i)
                    acc += gbuf[r * 8 + i][kidx] * w2[((size_t)r * 1024 + oP) * Rn + i];
                size_t idx = ((((size_t)b * 16 + (m >> 4)) * 9 + kidx) * 32 + lane) * 16 + slot;
                kern_frag[idx] = (__bf16)acc;
            }
        }
    }
}

// =====================================================================
// k2: guide classifiers + first-occurrence argmax -> region bytes
// =====================================================================
__global__ void k2_guide(const float* __restrict__ guide,
                         const float* __restrict__ w_spa, const float* __restrict__ b_spa,
                         const float* __restrict__ w_spec, const float* __restrict__ b_spec,
                         unsigned char* __restrict__ rspa, unsigned char* __restrict__ rspec) {
    __shared__ float wsA[Rn][Cn], wsB[Rn][Cn];
    int t = threadIdx.x;
    if (t < Rn * Cn) { wsA[t / Cn][t % Cn] = w_spa[t]; wsB[t / Cn][t % Cn] = w_spec[t]; }
    __syncthreads();
    int p = blockIdx.x * blockDim.x + t;
    int b = p / HWn, hw = p % HWn;
    float gv[Cn];
    #pragma unroll
    for (int c = 0; c < Cn; ++c) gv[c] = guide[((size_t)b * Cn + c) * HWn + hw];
    float bestA = -1e30f, bestB = -1e30f; int ia = 0, ib = 0;
    #pragma unroll
    for (int r = 0; r < Rn; ++r) {
        float da = b_spa[r], db = b_spec[r];
        #pragma unroll
        for (int c = 0; c < Cn; ++c) { da += gv[c] * wsA[r][c]; db += gv[c] * wsB[r][c]; }
        if (da > bestA) { bestA = da; ia = r; }
        if (db > bestB) { bestB = db; ib = r; }
    }
    rspa[p] = (unsigned char)ia;
    rspec[p] = (unsigned char)ib;
}

// =====================================================================
// kP: transpose input f32 [b][c][h][w] -> inputT bf16 [b][h][w][c]
//     (channel-innermost: enables contiguous TDM tiles in k3)
// =====================================================================
__global__ void kP_transpose(const float* __restrict__ input, __bf16* __restrict__ inputT) {
    __shared__ float row[Cn][132];
    const int h = blockIdx.x, b = blockIdx.y, t = threadIdx.x;
    for (int idx = t; idx < Cn * Wn; idx += 256) {
        int c = idx >> 7, x = idx & 127;                     // coalesced read
        row[c][x] = input[(((size_t)b * Cn + c) * Hn + h) * Wn + x];
    }
    __syncthreads();
    for (int idx = t; idx < Cn * Wn; idx += 256) {
        int x = idx >> 5, c = idx & 31;                      // 64B-contig writes
        inputT[(((size_t)b * Hn + h) * Wn + x) * Cn + c] = (__bf16)row[c][x];
    }
}

// =====================================================================
// k3: correlation implicit GEMM (bf16 WMMA) + fused argmax selection.
//     grid=(H,B), block=256 (8 waves). Wave r owns region r (32 M-rows).
//     LDS tile2[ry][x(130)][c(32)] staged by one TDM 2D load from inputT.
//     Output featT[b][h][w][c2] channel-innermost -> 16B selection stores.
// =====================================================================
__global__ void __launch_bounds__(256)
k3_corr(const __bf16* __restrict__ inputT, const __bf16* __restrict__ kern_frag,
        const unsigned char* __restrict__ rspa, const unsigned char* __restrict__ rspec,
        __bf16* __restrict__ featT) {
    __shared__ __align__(32) __bf16 tile2[3][130][Cn];   // rows contiguous: stride 4160
    const int h = blockIdx.x, b = blockIdx.y, t = threadIdx.x;

    // tile origin: (y=h-1, x=-1, c=0); left/top halo wraps (borders only)
    const __bf16* src0 = inputT + ((long long)b * HWn + (long long)(h - 1) * Wn) * Cn - Cn;

#if USE_TDM
    if (t < 32) {
        tdm_load_2d((unsigned)(size_t)(void*)&tile2[0][0][0], (const void*)src0,
                    130 * Cn, Wn * Cn);
        __builtin_amdgcn_s_wait_tensorcnt(0);
    }
    __syncthreads();
#else
    for (int ry = 0; ry < 3; ++ry) {
        const __bf16* src = src0 + (size_t)ry * Wn * Cn;
        __bf16* dst = &tile2[ry][0][0];
        for (int ch = t; ch < 520; ch += 256)
            *(v8bf*)(dst + ch * 8) = *(const v8bf*)(src + (size_t)ch * 8);
    }
    __syncthreads();
#endif

    const int r = t >> 5, lane = t & 31;          // wave id == region
    const int n = lane & 15, khalf = lane >> 4;
    const __bf16* lanep = &tile2[0][0][0] + (n * Cn + khalf * 16);

    // preload all 18 A fragments for this wave's 2 M-tiles
    const __bf16* Ab = kern_frag + (size_t)b * 16 * 9 * 32 * 16;
    v16bf afr[2][9];
    #pragma unroll
    for (int mt = 0; mt < 2; ++mt)
        #pragma unroll
        for (int kt = 0; kt < 9; ++kt)
            afr[mt][kt] = *(const v16bf*)(Ab + ((((size_t)r * 2 + mt) * 9 + kt) * 32 + lane) * 16);

    v8f acc[2][8];
    #pragma unroll
    for (int mt = 0; mt < 2; ++mt)
        #pragma unroll
        for (int nt = 0; nt < 8; ++nt) acc[mt][nt] = {};

    #pragma unroll
    for (int nt = 0; nt < 8; ++nt) {
        #pragma unroll
        for (int kt = 0; kt < 9; ++kt) {          // kt == kernel position q
            const int ky = kt / 3, kx = kt % 3;
            v16bf bf = *(const v16bf*)(lanep + (ky * 130 + nt * 16 + kx) * Cn);
            #pragma unroll
            for (int mt = 0; mt < 2; ++mt)
                acc[mt][nt] = __builtin_amdgcn_wmma_f32_16x16x32_bf16(
                    false, afr[mt][kt], false, bf, (short)0, acc[mt][nt], false, false);
        }
    }

    // fused selection: 16B stores of 8 consecutive channels (featT ch-inner)
    const int rowoff = khalf * 8;
    #pragma unroll
    for (int nt = 0; nt < 8; ++nt) {
        int px = nt * 16 + n;
        size_t pidx = ((size_t)b * Hn + h) * Wn + px;
        int rs = rspa[pidx], rc = rspec[pidx];
        __bf16* fp = featT + pidx * 64;
        if (rs == r) {
            #pragma unroll
            for (int mt = 0; mt < 2; ++mt) {
                v8bf pk;
                #pragma unroll
                for (int v = 0; v < 8; ++v) pk[v] = (__bf16)acc[mt][nt][v];
                *(v8bf*)(fp + mt * 16 + rowoff) = pk;
            }
        }
        if (rc == r) {
            #pragma unroll
            for (int mt = 0; mt < 2; ++mt) {
                v8bf pk;
                #pragma unroll
                for (int v = 0; v < 8; ++v) pk[v] = (__bf16)acc[mt][nt][v];
                *(v8bf*)(fp + 32 + mt * 16 + rowoff) = pk;
            }
        }
    }
}

// =====================================================================
// k4: 3x3 fusion conv (bf16 WMMA) + bias + residual -> f32 output.
//     grid=(H,B), block=256. Wave nt owns one 16-pixel N-tile.
//     LDS ftile[ry][x(130)][c2(64)] staged by one TDM 2D load from featT.
// =====================================================================
__global__ void __launch_bounds__(256)
k4_fuse(const __bf16* __restrict__ featT, const __bf16* __restrict__ wf_frag,
        const float* __restrict__ b_f, const float* __restrict__ input,
        float* __restrict__ out) {
    __shared__ __align__(32) __bf16 ftile[3][130][64];   // rows contiguous: stride 8320
    const int h = blockIdx.x, b = blockIdx.y, t = threadIdx.x;

    const __bf16* src0 = featT + ((long long)b * HWn + (long long)(h - 1) * Wn) * 64 - 64;

#if USE_TDM
    if (t < 32) {
        tdm_load_2d((unsigned)(size_t)(void*)&ftile[0][0][0], (const void*)src0,
                    130 * 64, Wn * 64);
        __builtin_amdgcn_s_wait_tensorcnt(0);
    }
    __syncthreads();
#else
    for (int ry = 0; ry < 3; ++ry) {
        const __bf16* src = src0 + (size_t)ry * Wn * 64;
        __bf16* dst = &ftile[ry][0][0];
        for (int ch = t; ch < 1040; ch += 256)
            *(v8bf*)(dst + ch * 8) = *(const v8bf*)(src + (size_t)ch * 8);
    }
    __syncthreads();
#endif

    const int nt = t >> 5, lane = t & 31;
    const int n = lane & 15, khalf = lane >> 4;
    const __bf16* lanep = &ftile[0][0][0] + ((nt * 16 + n) * 64 + khalf * 16);

    v8f acc[2];
    acc[0] = {}; acc[1] = {};

    #pragma unroll
    for (int kt = 0; kt < 18; ++kt) {
        const int q = kt >> 1, half = kt & 1;
        const int ky = q / 3, kx = q % 3;
        v16bf bf = *(const v16bf*)(lanep + (ky * 130 + kx) * 64 + half * 32);
        #pragma unroll
        for (int mt = 0; mt < 2; ++mt) {
            v16bf af = *(const v16bf*)(wf_frag + (((size_t)mt * 18 + kt) * 32 + lane) * 16);
            acc[mt] = __builtin_amdgcn_wmma_f32_16x16x32_bf16(
                false, af, false, bf, (short)0, acc[mt], false, false);
        }
    }

    const int rowoff = khalf * 8;
    const int px = nt * 16 + n;
    #pragma unroll
    for (int mt = 0; mt < 2; ++mt)
        #pragma unroll
        for (int v = 0; v < 8; ++v) {
            int o = mt * 16 + rowoff + v;
            size_t gi = (((size_t)b * OUTn + o) * Hn + h) * Wn + px;
            out[gi] = acc[mt][v] + b_f[o] + input[gi];
        }
}

// =====================================================================
extern "C" void kernel_launch(void* const* d_in, const int* in_sizes, int n_in,
                              void* d_out, int out_size, void* d_ws, size_t ws_size,
                              hipStream_t stream) {
    const float* input  = (const float*)d_in[0];
    const float* guide  = (const float*)d_in[1];
    const float* w1     = (const float*)d_in[2];
    const float* b1     = (const float*)d_in[3];
    const float* w2     = (const float*)d_in[4];
    const float* b2     = (const float*)d_in[5];
    const float* w_spa  = (const float*)d_in[6];
    const float* b_spa  = (const float*)d_in[7];
    const float* w_spec = (const float*)d_in[8];
    const float* b_spec = (const float*)d_in[9];
    const float* w_f    = (const float*)d_in[10];
    const float* b_f    = (const float*)d_in[11];
    float* out = (float*)d_out;

    char* ws = (char*)d_ws;
    __bf16* kern_frag = (__bf16*)(ws);                    // 2,359,296 B
    __bf16* wf_frag   = (__bf16*)(ws + 2359296);          //    36,864 B
    unsigned char* rspa  = (unsigned char*)(ws + 2396160);//   262,144 B
    unsigned char* rspec = (unsigned char*)(ws + 2658304);//   262,144 B
    __bf16* inputT    = (__bf16*)(ws + 2920448);          // 16,777,216 B  [b][h][w][c]
    __bf16* featT     = (__bf16*)(ws + 19697664);         // 33,554,432 B  [b][h][w][c2]

    k0_pack_wf<<<1, 256, 0, stream>>>(w_f, wf_frag);
    k1_gen<<<Bn, 288, 0, stream>>>(input, w1, b1, w2, b2, kern_frag);
    k2_guide<<<(Bn * HWn) / 256, 256, 0, stream>>>(guide, w_spa, b_spa, w_spec, b_spec,
                                                   rspa, rspec);
    kP_transpose<<<dim3(Hn, Bn), 256, 0, stream>>>(input, inputT);
    k3_corr<<<dim3(Hn, Bn), 256, 0, stream>>>(inputT, kern_frag, rspa, rspec, featT);
    k4_fuse<<<dim3(Hn, Bn), 256, 0, stream>>>(featT, wf_frag, b_f, input, out);
}
